// PyTorchManualGroupedLinear_65395172049358
// MI455X (gfx1250) — matrix operand order repaired and checked by reading, not compiled, not run on hardware
//
#include <hip/hip_runtime.h>
#include <hip/hip_bf16.h>

#define NUM_EXPERTS 8
#define KDIM 2048
#define NDIM 2048
#define TOTAL 32768
#define TM 128           // M tile (tokens) per block
#define TN 256           // N tile (out features) per block
#define KB 64            // K chunk per LDS stage (2 WMMA k-steps)
#define LDA 72           // padded bf16 row length: 144B stride -> conflict-free frag loads
#define MAX_MT (TOTAL / TM + NUM_EXPERTS)   // 264 worst-case M tiles

typedef float  v8f   __attribute__((ext_vector_type(8)));
typedef __bf16 v4bf  __attribute__((ext_vector_type(4)));
typedef __bf16 v8bf  __attribute__((ext_vector_type(8)));
typedef __bf16 v16bf __attribute__((ext_vector_type(16)));

static __device__ __forceinline__ v8f wmma_bf16(v16bf a, v16bf b, v8f c) {
  // D = A(16x32 bf16) * B(32x16 bf16) + C(16x16 f32)
  return __builtin_amdgcn_wmma_f32_16x16x32_bf16(false, a, false, b, (short)0, c,
                                                 false, false);
}

// ---------------- kernel 1: zero counters/cursors ----------------
__global__ void k_zero(int* __restrict__ counts, int* __restrict__ cursors) {
  int t = threadIdx.x;
  if (t < NUM_EXPERTS) { counts[t] = 0; cursors[t] = 0; }
}

// ---------------- kernel 2: expert histogram ----------------
__global__ void k_hist(const int* __restrict__ assign, int* __restrict__ counts) {
  __shared__ int bc[NUM_EXPERTS];
  int tid = threadIdx.x;
  if (tid < NUM_EXPERTS) bc[tid] = 0;
  __syncthreads();
  int t = blockIdx.x * blockDim.x + tid;
  if (t < TOTAL) atomicAdd(&bc[assign[t]], 1);
  __syncthreads();
  if (tid < NUM_EXPERTS && bc[tid] != 0) atomicAdd(&counts[tid], bc[tid]);
}

// ---------------- kernel 3: scan + tile table (1 thread) ----------------
__global__ void k_scan(const int* __restrict__ counts, int* __restrict__ offsets,
                       int* __restrict__ tiletab, int* __restrict__ numt) {
  int off = 0, nt = 0;
  for (int e = 0; e < NUM_EXPERTS; ++e) {
    offsets[e] = off;
    int c = counts[e];
    for (int r = 0; r < c; r += TM) {
      tiletab[nt * 3 + 0] = e;
      tiletab[nt * 3 + 1] = off + r;
      tiletab[nt * 3 + 2] = (c - r < TM) ? (c - r) : TM;
      ++nt;
    }
    off += c;
  }
  offsets[NUM_EXPERTS] = off;
  *numt = nt;
}

// ---------------- kernel 4: scatter tokens into expert-contiguous order ----
__global__ void k_scatter(const int* __restrict__ assign, const int* __restrict__ offsets,
                          int* __restrict__ cursors, int* __restrict__ sorted) {
  int t = blockIdx.x * blockDim.x + threadIdx.x;
  if (t < TOTAL) {
    int e = assign[t];
    int pos = offsets[e] + atomicAdd(&cursors[e], 1);
    sorted[pos] = t;
  }
}

// ---------------- kernel 5: grouped WMMA GEMM (128x256 tile, 64x64/wave) ----
__launch_bounds__(256, 1)
__global__ void k_gemm(const float* __restrict__ x, const float* __restrict__ w,
                       const int* __restrict__ sorted, const int* __restrict__ tiletab,
                       const int* __restrict__ numt, float* __restrict__ out) {
  __shared__ __align__(16) __bf16 As[2][TM][LDA];
  __shared__ __align__(16) __bf16 Bs[2][TN][LDA];
  __shared__ int toks[TM];

  const int mt = blockIdx.y;            // M-tile index (expert-sorted tokens)
  if (mt >= *numt) return;
  const int e      = tiletab[mt * 3 + 0];
  const int rstart = tiletab[mt * 3 + 1];
  const int nrows  = tiletab[mt * 3 + 2];
  const int n0     = blockIdx.x * TN;   // N-tile fastest -> A tile reused via L2

  const int tid  = threadIdx.x;
  const int lane = tid & 31;
  const int wave = tid >> 5;
  const int wm   = (wave & 1) * 64;     // wave's 64 rows in the 128-row tile
  const int wn   = (wave >> 1) * 64;    // wave's 64 cols in the 256-col tile

  if (tid < TM) toks[tid] = (tid < nrows) ? sorted[rstart + tid] : -1;
  __syncthreads();

  const float* __restrict__ wb = w + (size_t)e * KDIM * NDIM;

  // ---- hoisted loader state: fixed (row, c4) slots per thread ----
  const int r0 = tid >> 4;              // 0..15; A rows r0+16i (i<8), B rows r0+16i (i<16)
  const int c4 = (tid & 15) * 4;        // float4 column within the K chunk
  const float* aptr[8];
  bool avalid[8];
#pragma unroll
  for (int i = 0; i < 8; ++i) {
    const int tk = toks[r0 + i * 16];
    avalid[i] = (tk >= 0);
    // invalid rows read row 0 (always safe) and are zero-selected post-load
    aptr[i] = x + (size_t)(avalid[i] ? tk : 0) * KDIM + c4;
  }
  const float* bptr = wb + (size_t)(n0 + r0) * KDIM + c4;  // + i*16*KDIM per slot

  float4 areg[8], breg[16];
  auto issue_loads = [&](int kc) {      // global -> registers (batched clauses)
    const int kk = kc * KB;
#pragma unroll
    for (int i = 0; i < 8; ++i) areg[i] = *(const float4*)(aptr[i] + kk);
#pragma unroll
    for (int i = 0; i < 16; ++i)
      breg[i] = *(const float4*)(bptr + (size_t)i * 16 * KDIM + kk);
  };
  auto convert_store = [&](int buf) {   // registers -> bf16 -> LDS
#pragma unroll
    for (int i = 0; i < 8; ++i) {
      const int r = r0 + i * 16;
      const float ax = avalid[i] ? areg[i].x : 0.f;
      const float ay = avalid[i] ? areg[i].y : 0.f;
      const float az = avalid[i] ? areg[i].z : 0.f;
      const float aw = avalid[i] ? areg[i].w : 0.f;
      v4bf ah = {(__bf16)ax, (__bf16)ay, (__bf16)az, (__bf16)aw};
      *(v4bf*)(&As[buf][r][c4]) = ah;
    }
#pragma unroll
    for (int i = 0; i < 16; ++i) {
      const int r = r0 + i * 16;
      v4bf bh = {(__bf16)breg[i].x, (__bf16)breg[i].y, (__bf16)breg[i].z,
                 (__bf16)breg[i].w};
      *(v4bf*)(&Bs[buf][r][c4]) = bh;
    }
  };

  // ---- fragment loads straight from row-major LDS per ISA VGPR layouts ----
  const int frow  = lane & 15;
  const int ka_hi = (lane >> 4) << 3;   // A frag: K base 0 / 8
  const int kb_hi = (lane >> 4) << 3;   // B frag: K base 0 / 8
  auto afrag = [&](int buf, int row0, int kk) -> v16bf {
    const __bf16* p = &As[buf][row0 + frow][kk + ka_hi];
    v8bf lo = *(const v8bf*)p;          // K = base+0..7
    v8bf hi = *(const v8bf*)(p + 16);   // K = base+16..23
    return __builtin_shufflevector(lo, hi, 0, 1, 2, 3, 4, 5, 6, 7,
                                   8, 9, 10, 11, 12, 13, 14, 15);
  };
  auto bfrag = [&](int buf, int col0, int kk) -> v16bf {
    const __bf16* p = &Bs[buf][col0 + frow][kk + kb_hi];
    v8bf lo = *(const v8bf*)p;          // K = base+0..7
    v8bf hi = *(const v8bf*)(p + 16);   // K = base+16..23
    return __builtin_shufflevector(lo, hi, 0, 1, 2, 3, 4, 5, 6, 7,
                                   8, 9, 10, 11, 12, 13, 14, 15);
  };

  v8f acc[4][4] = {};                   // 16 x (16x16 f32) accumulators per wave

  auto compute = [&](int buf) {
#pragma unroll
    for (int ks = 0; ks < KB; ks += 32) {
      v16bf a[4], b[4];
#pragma unroll
      for (int im = 0; im < 4; ++im) a[im] = afrag(buf, wm + im * 16, ks);
#pragma unroll
      for (int jn = 0; jn < 4; ++jn) b[jn] = bfrag(buf, wn + jn * 16, ks);
#pragma unroll
      for (int im = 0; im < 4; ++im)
#pragma unroll
        for (int jn = 0; jn < 4; ++jn)
          acc[im][jn] = wmma_bf16(a[im], b[jn], acc[im][jn]);
    }
  };

  // ---- software pipeline: loads(k+1) || wmma(k) ----
  issue_loads(0);
  convert_store(0);
  __syncthreads();

  const int NC = KDIM / KB;             // 32 chunks
  for (int kc = 0; kc < NC; ++kc) {
    const int buf = kc & 1;
    if (kc + 1 < NC) issue_loads(kc + 1);   // in flight during the WMMAs below
    compute(buf);
    if (kc + 1 < NC) convert_store(buf ^ 1);
    __syncthreads();
  }

  // epilogue: C/D layout -> lanes 0-15: M=r, lanes 16-31: M=r+8; N = lane%16
  const int rsel = (lane >> 4) << 3;
#pragma unroll
  for (int im = 0; im < 4; ++im)
#pragma unroll
    for (int j = 0; j < 4; ++j)
#pragma unroll
      for (int r = 0; r < 8; ++r) {
        const int lrow = wm + im * 16 + rsel + r;
        if (lrow < nrows) {
          const int col = n0 + wn + j * 16 + frow;
          // streaming store: keep the 256MB output out of L2 (weights live there)
          __builtin_nontemporal_store(acc[im][j][r],
                                      out + (size_t)toks[lrow] * NDIM + col);
        }
      }
}

extern "C" void kernel_launch(void* const* d_in, const int* in_sizes, int n_in,
                              void* d_out, int out_size, void* d_ws, size_t ws_size,
                              hipStream_t stream) {
  (void)in_sizes; (void)n_in; (void)out_size; (void)ws_size;
  const float* x      = (const float*)d_in[0];
  const int*   assign = (const int*)d_in[1];
  const float* w      = (const float*)d_in[2];
  float*       out    = (float*)d_out;

  int* ws      = (int*)d_ws;
  int* counts  = ws + 0;      // 8
  int* cursors = ws + 8;      // 8
  int* offsets = ws + 16;     // 9
  int* numt    = ws + 25;     // 1
  int* tiletab = ws + 32;     // MAX_MT * 3 = 792 ints
  int* sorted  = ws + 1024;   // 32768 ints

  k_zero<<<1, 32, 0, stream>>>(counts, cursors);
  k_hist<<<TOTAL / 256, 256, 0, stream>>>(assign, counts);
  k_scan<<<1, 1, 0, stream>>>(counts, offsets, tiletab, numt);
  k_scatter<<<TOTAL / 256, 256, 0, stream>>>(assign, offsets, cursors, sorted);

  dim3 grid(NDIM / TN, MAX_MT);   // x = N-tiles (fast) for L2 reuse of A tile
  k_gemm<<<grid, 256, 0, stream>>>(x, w, sorted, tiletab, numt, out);
}
